// GCN_nc_45561013076147
// MI455X (gfx1250) — matrix operand order repaired and checked by reading, not compiled, not run on hardware
//
#include <hip/hip_runtime.h>
#include <hip/hip_bf16.h>
#include <math.h>

// ---------------------------------------------------------------------------
// GCN encoder + dot-product decoder for MI455X (gfx1250, wave32, WMMA).
//
// Pipeline:
//   1. x (f32) -> xb (bf16);  W1 -> W1t (bf16, transposed N x K);  W2 -> W2t
//   2. S1 = xb @ W1    : v_wmma_f32_16x16x32_bf16, f32 accumulate
//   3. h1[src] += S1[dst] * w   (float4 gather + global_atomic_add_f32)
//   4. h1 -> h1b (bf16); S2 = h1b @ W2 (WMMA); h2[src] += S2[dst]*w
//   5. z = h2 / max(||h2||, 1e-12)  (one wave32 per row, shfl_xor reduce)
//      -> z (f32, to d_out) and zb (bf16, for decode)
//   6. A_pred = sigmoid(zb @ zb^T)  : WMMA, K=64 = 2 k-steps, nontemporal
//      f32 stores (256 MB output must stream past L2).
// ---------------------------------------------------------------------------

typedef __attribute__((ext_vector_type(16))) __bf16 v16bf;
typedef __attribute__((ext_vector_type(8)))  __bf16 v8bf;
typedef __attribute__((ext_vector_type(8)))  float  v8f;

static constexpr int   NN   = 8192;
static constexpr int   FIN  = 512;
static constexpr int   HID  = 256;
static constexpr int   EMB  = 64;
static constexpr int   NE   = 262144;

// ---- WMMA fragment loaders ------------------------------------------------
// A-matrix 16x32 bf16 (ISA 7.12.2, 16-bit A layout):
//   lanes 0-15 : row M = lane,      halfs[0..7] = K 0..7,  halfs[8..15] = K 16..23
//   lanes 16-31: row M = lane-16,   halfs[0..7] = K 8..15, halfs[8..15] = K 24..31
__device__ __forceinline__ v16bf load_frag_a(const __bf16* A, int mBase, int ld,
                                             int kBase, int lane) {
    const int m  = lane & 15;
    const int kh = (lane >> 4) * 8;
    const __bf16* p = A + (size_t)(mBase + m) * ld + kBase + kh;
    v8bf lo = *(const v8bf*)(p);        // K kh .. kh+7
    v8bf hi = *(const v8bf*)(p + 16);   // K kh+16 .. kh+23
    v16bf r;
#pragma unroll
    for (int i = 0; i < 8; ++i) { r[i] = lo[i]; r[i + 8] = hi[i]; }
    return r;
}

// B-matrix 32x16 bf16, sourced from a transposed (N x K row-major) array so
// every lane reads two runs of 8 contiguous K values:
//   lanes 0-15 : col N = lane,      halfs[0..7] = K 0..7,  halfs[8..15] = K 16..23
//   lanes 16-31: col N = lane-16,   halfs[0..7] = K 8..15, halfs[8..15] = K 24..31
__device__ __forceinline__ v16bf load_frag_bt(const __bf16* Bt, int nBase, int ld,
                                              int kBase, int lane) {
    const int n  = lane & 15;
    const int k0 = kBase + (lane >> 4) * 8;
    const __bf16* p = Bt + (size_t)(nBase + n) * ld + k0;
    v8bf lo = *(const v8bf*)(p);
    v8bf hi = *(const v8bf*)(p + 16);
    v16bf r;
#pragma unroll
    for (int i = 0; i < 8; ++i) { r[i] = lo[i]; r[i + 8] = hi[i]; }
    return r;
}

// ---- conversion / init kernels -------------------------------------------
__global__ void f32_to_bf16_kernel(const float* __restrict__ in,
                                   __bf16* __restrict__ out, int n) {
    int i = blockIdx.x * blockDim.x + threadIdx.x;
    if (i < n) out[i] = (__bf16)in[i];
}

// W (K x N, row-major f32) -> Wt (N x K, row-major bf16)
__global__ void transpose_to_bf16_kernel(const float* __restrict__ W,
                                         __bf16* __restrict__ Wt, int K, int Nc) {
    int i = blockIdx.x * blockDim.x + threadIdx.x;
    if (i < K * Nc) {
        int k = i / Nc, n = i % Nc;
        Wt[(size_t)n * K + k] = (__bf16)W[i];
    }
}

__global__ void zero_kernel(float* __restrict__ p, int n) {
    int i = blockIdx.x * blockDim.x + threadIdx.x;
    if (i < n) p[i] = 0.0f;
}

// ---- GEMM: C(MxN,f32) = A(MxK,bf16) @ Bt(NxK,bf16)^T ----------------------
__global__ __launch_bounds__(256) void gemm_bf16_kernel(
    const __bf16* __restrict__ A, const __bf16* __restrict__ Bt,
    float* __restrict__ C, int M, int N, int K) {
    const int lane   = threadIdx.x & 31;
    const int waveId = blockIdx.x * (blockDim.x >> 5) + (threadIdx.x >> 5);
    const int tilesN = N >> 4;
    const int mBase  = (waveId / tilesN) << 4;
    const int nBase  = (waveId % tilesN) << 4;
    if (mBase >= M) return;

    v8f c = {};
    for (int k = 0; k < K; k += 32) {
        v16bf a = load_frag_a(A, mBase, K, k, lane);
        v16bf b = load_frag_bt(Bt, nBase, K, k, lane);
        c = __builtin_amdgcn_wmma_f32_16x16x32_bf16(false, a, false, b,
                                                    (short)0, c, false, false);
    }
    // C/D layout: VGPR r -> M = mBase + r + (lane/16)*8, N = nBase + (lane&15)
    const int n    = nBase + (lane & 15);
    const int mOff = mBase + ((lane >> 4) * 8);
#pragma unroll
    for (int r = 0; r < 8; ++r)
        C[(size_t)(mOff + r) * N + n] = c[r];
}

// ---- sparse aggregation: out[src[e]] += S[dst[e]] * w[e] ------------------
__global__ __launch_bounds__(256) void agg_kernel(
    const float* __restrict__ S, const int* __restrict__ src,
    const int* __restrict__ dst, const float* __restrict__ w,
    float* __restrict__ out, int E, int F) {
    const int chunks = F >> 2;                      // float4 per thread
    int t = blockIdx.x * blockDim.x + threadIdx.x;
    if (t >= E * chunks) return;
    const int e = t / chunks, c = (t % chunks) << 2;
    const float  ww = w[e];
    const float4 v  = *(const float4*)(S + (size_t)dst[e] * F + c);
    float* o = out + (size_t)src[e] * F + c;
    atomicAdd(o + 0, v.x * ww);
    atomicAdd(o + 1, v.y * ww);
    atomicAdd(o + 2, v.z * ww);
    atomicAdd(o + 3, v.w * ww);
}

// ---- row L2-normalize: one wave32 per 64-wide row -------------------------
__global__ __launch_bounds__(256) void normalize_kernel(
    const float* __restrict__ h, float* __restrict__ zOut,
    __bf16* __restrict__ zb) {
    const int row  = blockIdx.x * 8 + (threadIdx.x >> 5);
    const int lane = threadIdx.x & 31;
    const float2 v = *(const float2*)(h + (size_t)row * EMB + lane * 2);
    float s = v.x * v.x + v.y * v.y;
#pragma unroll
    for (int off = 16; off > 0; off >>= 1) s += __shfl_xor(s, off, 32);
    const float inv = 1.0f / fmaxf(sqrtf(s), 1e-12f);
    const float zx = v.x * inv, zy = v.y * inv;
    *(float2*)(zOut + (size_t)row * EMB + lane * 2) = make_float2(zx, zy);
    zb[(size_t)row * EMB + lane * 2 + 0] = (__bf16)zx;
    zb[(size_t)row * EMB + lane * 2 + 1] = (__bf16)zy;
}

// ---- decode: A_pred = sigmoid(zb @ zb^T), streamed with NT stores ---------
// Each wave computes a 16x64 strip: A fragments (both K-halves) loaded once,
// reused across 4 B tiles. zb is 1 MB -> L2 resident; output streams.
__global__ __launch_bounds__(256) void decode_kernel(
    const __bf16* __restrict__ zb, float* __restrict__ out) {
    const int lane   = threadIdx.x & 31;
    const int waveId = blockIdx.x * 8 + (threadIdx.x >> 5);
    const int nGroups = (NN / 16) / 4;              // 128 strips of 64 cols
    const int mBase  = (waveId / nGroups) * 16;
    const int nBase  = (waveId % nGroups) * 64;

    const v16bf a0 = load_frag_a(zb, mBase, EMB, 0,  lane);
    const v16bf a1 = load_frag_a(zb, mBase, EMB, 32, lane);

#pragma unroll
    for (int t = 0; t < 4; ++t) {
        const int nb = nBase + t * 16;
        v16bf b0 = load_frag_bt(zb, nb, EMB, 0,  lane);
        v16bf b1 = load_frag_bt(zb, nb, EMB, 32, lane);
        v8f c = {};
        c = __builtin_amdgcn_wmma_f32_16x16x32_bf16(false, a0, false, b0,
                                                    (short)0, c, false, false);
        c = __builtin_amdgcn_wmma_f32_16x16x32_bf16(false, a1, false, b1,
                                                    (short)0, c, false, false);
        const int n    = nb + (lane & 15);
        const int mOff = mBase + ((lane >> 4) * 8);
#pragma unroll
        for (int r = 0; r < 8; ++r) {
            const float sg = 1.0f / (1.0f + __expf(-c[r]));
            __builtin_nontemporal_store(sg, out + (size_t)(mOff + r) * NN + n);
        }
    }
}

// ---------------------------------------------------------------------------
extern "C" void kernel_launch(void* const* d_in, const int* in_sizes, int n_in,
                              void* d_out, int out_size, void* d_ws, size_t ws_size,
                              hipStream_t stream) {
    const float* x   = (const float*)d_in[0];       // [N, FIN]
    const int*   ei  = (const int*)  d_in[1];       // [2, E] : row0=src, row1=dst
    const float* ew  = (const float*)d_in[2];       // [E]
    const float* W1  = (const float*)d_in[3];       // [FIN, HID]
    const float* W2  = (const float*)d_in[4];       // [HID, EMB]
    const int* srcIdx = ei;
    const int* dstIdx = ei + NE;

    float* A_pred = (float*)d_out;                          // [N, N]
    float* zOut   = (float*)d_out + (size_t)NN * NN;        // [N, EMB]

    // workspace carve (256B-aligned regions)
    char*  base = (char*)d_ws;
    size_t off  = 0;
    auto alloc = [&](size_t bytes) -> void* {
        void* p = base + off;
        off += (bytes + 255) & ~(size_t)255;
        return p;
    };
    __bf16* xb  = (__bf16*)alloc((size_t)NN  * FIN * 2);
    __bf16* W1t = (__bf16*)alloc((size_t)HID * FIN * 2);
    float*  S1  = (float*) alloc((size_t)NN  * HID * 4);
    float*  h1  = (float*) alloc((size_t)NN  * HID * 4);
    __bf16* h1b = (__bf16*)alloc((size_t)NN  * HID * 2);
    __bf16* W2t = (__bf16*)alloc((size_t)EMB * HID * 2);
    float*  S2  = (float*) alloc((size_t)NN  * EMB * 4);
    float*  h2  = (float*) alloc((size_t)NN  * EMB * 4);
    __bf16* zb  = (__bf16*)alloc((size_t)NN  * EMB * 2);

    const int B = 256;

    // 1. precision conversion + weight transposes
    f32_to_bf16_kernel<<<(NN * FIN + B - 1) / B, B, 0, stream>>>(x, xb, NN * FIN);
    transpose_to_bf16_kernel<<<(FIN * HID + B - 1) / B, B, 0, stream>>>(W1, W1t, FIN, HID);
    transpose_to_bf16_kernel<<<(HID * EMB + B - 1) / B, B, 0, stream>>>(W2, W2t, HID, EMB);
    zero_kernel<<<(NN * HID + B - 1) / B, B, 0, stream>>>(h1, NN * HID);
    zero_kernel<<<(NN * EMB + B - 1) / B, B, 0, stream>>>(h2, NN * EMB);

    // 2. S1 = xb @ W1   (512 x 16 tiles = 8192 waves, 8 waves/block)
    gemm_bf16_kernel<<<(NN / 16) * (HID / 16) / 8, B, 0, stream>>>(xb, W1t, S1, NN, HID, FIN);

    // 3. h1[src] += S1[dst] * w
    agg_kernel<<<(NE * (HID / 4) + B - 1) / B, B, 0, stream>>>(S1, srcIdx, dstIdx, ew, h1, NE, HID);

    // 4. S2 = h1b @ W2 ; h2[src] += S2[dst] * w
    f32_to_bf16_kernel<<<(NN * HID + B - 1) / B, B, 0, stream>>>(h1, h1b, NN * HID);
    gemm_bf16_kernel<<<(NN / 16) * (EMB / 16) / 8, B, 0, stream>>>(h1b, W2t, S2, NN, EMB, HID);
    agg_kernel<<<(NE * (EMB / 4) + B - 1) / B, B, 0, stream>>>(S2, srcIdx, dstIdx, ew, h2, NE, EMB);

    // 5. z = normalize(h2)  (8192 rows, 8 waves/block)
    normalize_kernel<<<NN / 8, B, 0, stream>>>(h2, zOut, zb);

    // 6. A_pred = sigmoid(z @ z^T)  (65536 waves, 8/block)
    decode_kernel<<<(NN / 16) * ((NN / 16) / 4) / 8, B, 0, stream>>>(zb, A_pred);
}
